// Induc_44066364457491
// MI455X (gfx1250) — matrix operand order repaired and checked by reading, not compiled, not run on hardware
//
#include <hip/hip_runtime.h>

// ---------------------------------------------------------------------------
// MI455X (gfx1250, wave32) fused sector-attention pipeline.
// All matmuls via v_wmma_f32_16x16x32_f16 (f16 operands, f32 accumulate).
// LDS operand tiles are fragment-major: each 16x32 A / 32x16 B fragment is
// 32 lanes x 16 contiguous halves -> fragment load = 2x ds_load_b128.
// Staging: thread t loads 16 contiguous floats (4x global_load_b128, clause)
// of row t>>2; tail chunks take a uniform scalar-branch slow path.
// ---------------------------------------------------------------------------

#define LDF 68                 // f32 scores scratch stride (floats); row = 272B (16B aligned)
#define NEG_BIG (-3.402823466e38f)

typedef __attribute__((ext_vector_type(16))) _Float16 v16h;
typedef __attribute__((ext_vector_type(8)))  _Float16 v8h;
typedef __attribute__((ext_vector_type(8)))  float    v8f;

static __device__ __forceinline__ v8f wmma16(v16h a, v16h b, v8f c) {
  return __builtin_amdgcn_wmma_f32_16x16x32_f16(
      /*neg_a=*/false, a, /*neg_b=*/false, b,
      /*c_mod=*/(short)0, c, /*reuse_a=*/false, /*reuse_b=*/false);
}

// ---- fragment-major addressing -------------------------------------------
// One fragment = 32 lanes * 16 halves = 512 halves (1 KB), 32B per lane.

// Flat half-index for element (k,n) of a 64x64 B matrix (B frag = kc*4+nt).
static __device__ __forceinline__ int b_addr(int k, int n) {
  int frag = ((k >> 5) << 2) | (n >> 4);
  int kl = k & 31;
  int lane = (n & 15) | (((kl >> 3) & 1) << 4);
  return ((frag << 5) + lane) * 16 + ((kl & 7) | ((kl & 16) >> 1));
}
// Flat half-index for element (m,k) of a 64x64 A matrix (A frag = mt*2+kc).
static __device__ __forceinline__ int a_addr(int m, int k) {
  int kc = k >> 5, kl = k & 31;
  int frag = ((m >> 4) << 1) | kc;
  int lane = (m & 15) | (((kl >> 3) & 1) << 4);
  int i = (kl & 7) | ((kl & 16) >> 1);
  return ((frag << 5) + lane) * 16 + i;
}
// Flat half-index for element (m,k) of a 16x64 A tile (frag = kc).
static __device__ __forceinline__ int a16_addr(int m, int k) {
  int kc = k >> 5, kl = k & 31;
  int lane = (m & 15) | (((kl >> 3) & 1) << 4);
  int i = (kl & 7) | ((kl & 16) >> 1);
  return ((kc << 5) + lane) * 16 + i;
}

// Contiguous per-lane fragment load: 32 bytes -> 2x ds_load_b128.
static __device__ __forceinline__ v16h load_frag(const _Float16* buf, int frag,
                                                 int lane) {
  return *(const v16h*)(buf + (((frag << 5) + lane) << 4));
}

// ---- staging helpers ------------------------------------------------------
// Unconditional: 16 contiguous floats of row e, cols c0..c0+15 (4x b128).
static __device__ __forceinline__ void load_row16_nc(const float* __restrict__ src,
                                                     long e, int c0, float* v) {
  const float4* p = (const float4*)(src + e * 64 + c0);
  float4 q0 = p[0], q1 = p[1], q2 = p[2], q3 = p[3];
  v[0] = q0.x;  v[1] = q0.y;  v[2]  = q0.z;  v[3]  = q0.w;
  v[4] = q1.x;  v[5] = q1.y;  v[6]  = q1.z;  v[7]  = q1.w;
  v[8] = q2.x;  v[9] = q2.y;  v[10] = q2.z;  v[11] = q2.w;
  v[12] = q3.x; v[13] = q3.y; v[14] = q3.z;  v[15] = q3.w;
}
// Guarded (tail chunks only): zero-fill rows past E.
static __device__ __forceinline__ void load_row16(const float* __restrict__ src,
                                                  long e, int E, int c0,
                                                  float* v) {
  if (e < E) {
    load_row16_nc(src, e, c0, v);
  } else {
#pragma unroll
    for (int j = 0; j < 16; ++j) v[j] = 0.f;
  }
}
// B natural: B[k=row][n=col]  (16x b16 scatter, stride 32B)
static __device__ __forceinline__ void put_b_nat(_Float16* dst, int r, int c0,
                                                 const float* v) {
#pragma unroll
  for (int j = 0; j < 16; ++j) dst[b_addr(r, c0 + j)] = (_Float16)v[j];
}
// B transposed: B[k=col][n=row]  (two contiguous 16B stores)
static __device__ __forceinline__ void put_b_tr(_Float16* dst, int r, int c0,
                                                const float* v) {
  v8h h0, h1;
#pragma unroll
  for (int j = 0; j < 8; ++j) { h0[j] = (_Float16)v[j]; h1[j] = (_Float16)v[8 + j]; }
  *(v8h*)(dst + b_addr(c0, r)) = h0;
  *(v8h*)(dst + b_addr(c0 + 8, r)) = h1;
}
// A transposed: A[m=col][k=row]  (16x b16 scatter)
static __device__ __forceinline__ void put_a_tr(_Float16* dst, int r, int c0,
                                                const float* v) {
#pragma unroll
  for (int j = 0; j < 16; ++j) dst[a_addr(c0 + j, r)] = (_Float16)v[j];
}
// A natural: A[m=row][k=col]  (two contiguous 16B stores)
static __device__ __forceinline__ void put_a_nat(_Float16* dst, int r, int c0,
                                                 const float* v) {
  v8h h0, h1;
#pragma unroll
  for (int j = 0; j < 8; ++j) { h0[j] = (_Float16)v[j]; h1[j] = (_Float16)v[8 + j]; }
  *(v8h*)(dst + a_addr(r, c0)) = h0;
  *(v8h*)(dst + a_addr(r, c0 + 8)) = h1;
}

// C/D (16x16 f32): lanes 0-15 -> N=lane, M=0..7 in c[0..7]; lanes 16-31 -> M=8..15.
static __device__ __forceinline__ void store_d_ldsf(float* t, int ld, int lane,
                                                    int n0, v8f c) {
  int n  = n0 + (lane & 15);
  int mb = (lane & 16) ? 8 : 0;
#pragma unroll
  for (int i = 0; i < 8; ++i) t[(mb + i) * ld + n] = c[i];
}
// Store a C/D tile column directly as f16 A-fragment data (16x64 tile).
static __device__ __forceinline__ void store_d_afrag(_Float16* buf, int lane,
                                                     int n0, v8f c) {
  int n  = n0 + (lane & 15);
  int mb = (lane & 16) ? 8 : 0;
#pragma unroll
  for (int i = 0; i < 8; ++i) buf[a16_addr(mb + i, n)] = (_Float16)c[i];
}

// A fragment built straight from global f32 row-major (ISA A layout).
static __device__ __forceinline__ v16h frag_a_gl(const float* base, int ld,
                                                 int lane, int k0) {
  int r  = lane & 15;
  int kh = k0 + ((lane & 16) ? 8 : 0);
  const float* p = base + (long)r * ld + kh;
  v16h a;
#pragma unroll
  for (int i = 0; i < 8; ++i) a[i] = (_Float16)p[i];
#pragma unroll
  for (int i = 0; i < 8; ++i) a[i + 8] = (_Float16)p[16 + i];
  return a;
}

static __device__ __forceinline__ unsigned f2ord(float f) {
  unsigned u = __float_as_uint(f);
  return (u & 0x80000000u) ? ~u : (u | 0x80000000u);
}
static __device__ __forceinline__ float ord2f(unsigned u) {
  return (u & 0x80000000u) ? __uint_as_float(u & 0x7fffffffu)
                           : __uint_as_float(~u);
}
static __device__ __forceinline__ float leaky(float v) {
  return v >= 0.f ? v : 0.01f * v;
}

// ---------------------------------------------------------------------------
// K0: zero/initialize workspace scalars.
__global__ void k_init(float* colsum, float* p1, unsigned* mEnc, float* zsum,
                       float* p2) {
  int t = threadIdx.x;
  if (t < 64) { colsum[t] = 0.f; zsum[t] = 0.f; mEnc[t] = f2ord(NEG_BIG); }
  for (int i = t; i < 4096; i += blockDim.x) { p1[i] = 0.f; p2[i] = 0.f; }
}

// K1: colsum[s] = sum_e ent2sec[e][s]
__global__ void k_colsum(const float* __restrict__ a, float* __restrict__ colsum,
                         int E) {
  __shared__ float part[4][64];
  int t = threadIdx.x;
  int s = t & 63, g = t >> 6;
  float acc = 0.f;
  for (long e = (long)blockIdx.x * 4 + g; e < E; e += (long)gridDim.x * 4)
    acc += a[e * 64 + s];
  part[g][s] = acc;
  __syncthreads();
  if (g == 0)
    atomicAdd(&colsum[s], part[0][s] + part[1][s] + part[2][s] + part[3][s]);
}

// K2: p1[s][d] += sum_e ent2sec[e][s] * x[e][d]
__global__ void k_sec1_accum(const float* __restrict__ adj,
                             const float* __restrict__ x,
                             float* __restrict__ p1, int E) {
  __shared__ alignas(32) _Float16 sAT[4096];  // A frags: ent2sec^T [s][e]
  __shared__ alignas(32) _Float16 sXN[4096];  // B frags: x [e][d]
  int t = threadIdx.x, lane = t & 31, w = t >> 5;
  int mt0 = w >> 2, nt = w & 3;               // wave owns tiles (mt0,nt),(mt0+2,nt)
  int rr = t >> 2, c0 = (t & 3) << 4;
  v8f acc0 = {}, acc1 = {};
  int nch = (E + 63) >> 6;
  for (int c = blockIdx.x; c < nch; c += gridDim.x) {
    long e0 = (long)c << 6;
    float va[16], vx[16];
    if (e0 + 64 <= E) {                        // uniform fast path
      load_row16_nc(adj, e0 + rr, c0, va);
      load_row16_nc(x, e0 + rr, c0, vx);
    } else {
      load_row16(adj, e0 + rr, E, c0, va);
      load_row16(x, e0 + rr, E, c0, vx);
    }
    put_a_tr(sAT, rr, c0, va);                // A[s=col][e=row]
    put_b_nat(sXN, rr, c0, vx);               // B[e=row][d=col]
    __syncthreads();
    acc0 = wmma16(load_frag(sAT, mt0 * 2 + 0, lane), load_frag(sXN, 0 + nt, lane), acc0);
    acc0 = wmma16(load_frag(sAT, mt0 * 2 + 1, lane), load_frag(sXN, 4 + nt, lane), acc0);
    acc1 = wmma16(load_frag(sAT, (mt0 + 2) * 2 + 0, lane), load_frag(sXN, 0 + nt, lane), acc1);
    acc1 = wmma16(load_frag(sAT, (mt0 + 2) * 2 + 1, lane), load_frag(sXN, 4 + nt, lane), acc1);
    __syncthreads();
  }
  int n = nt * 16 + (lane & 15);
  int mb = (lane & 16) ? 8 : 0;
#pragma unroll
  for (int i = 0; i < 8; ++i) {
    atomicAdd(&p1[(mt0 * 16 + mb + i) * 64 + n], acc0[i]);
    atomicAdd(&p1[((mt0 + 2) * 16 + mb + i) * 64 + n], acc1[i]);
  }
}

// K3: sector1[s][d] = p1[s][d] / colsum[s]
__global__ void k_sec1_fin(const float* p1, const float* colsum, float* sec1) {
  for (int i = threadIdx.x; i < 4096; i += blockDim.x)
    sec1[i] = p1[i] / colsum[i >> 6];
}

// K4: mEnc[s] = ordered max_e score[e][s];  score = sector1 . x^T (recomputed)
__global__ void k_colmax(const float* __restrict__ x,
                         const float* __restrict__ sec1,
                         unsigned* __restrict__ mEnc, int E) {
  __shared__ alignas(32) _Float16 sS1[4096];  // A frags: sector1 [s][d]
  __shared__ alignas(32) _Float16 sXT[4096];  // B frags: x^T [d][e]
  __shared__ float sSc[64 * LDF];             // scores^T chunk [s][e]
  int t = threadIdx.x, lane = t & 31, w = t >> 5;
  int mt0 = w >> 2, nt = w & 3;
  int rr = t >> 2, c0 = (t & 3) << 4;
  {
    float v[16];
    load_row16_nc(sec1, rr, c0, v);
    put_a_nat(sS1, rr, c0, v);
  }
  float runmax = NEG_BIG;
  __syncthreads();
  int nch = (E + 63) >> 6;
  for (int c = blockIdx.x; c < nch; c += gridDim.x) {
    long e0 = (long)c << 6;
    float v[16];
    if (e0 + 64 <= E) load_row16_nc(x, e0 + rr, c0, v);
    else              load_row16(x, e0 + rr, E, c0, v);
    put_b_tr(sXT, rr, c0, v);                 // B[d=col][e=row]
    __syncthreads();
#pragma unroll
    for (int tt = 0; tt < 2; ++tt) {
      int mt = mt0 + tt * 2;
      v8f sc = {};
      sc = wmma16(load_frag(sS1, mt * 2 + 0, lane), load_frag(sXT, 0 + nt, lane), sc);
      sc = wmma16(load_frag(sS1, mt * 2 + 1, lane), load_frag(sXT, 4 + nt, lane), sc);
      int nl = lane & 15;
      int mbase = mt * 16 + ((lane & 16) ? 8 : 0);
      bool valid = (e0 + nt * 16 + nl) < E;
#pragma unroll
      for (int i = 0; i < 8; ++i)
        sSc[(mbase + i) * LDF + nt * 16 + nl] = valid ? sc[i] : NEG_BIG;
    }
    __syncthreads();
    if (t < 64) {
      const float4* q = (const float4*)(sSc + t * LDF);
#pragma unroll
      for (int j = 0; j < 16; ++j) {
        float4 z = q[j];
        runmax = fmaxf(runmax, fmaxf(fmaxf(z.x, z.y), fmaxf(z.z, z.w)));
      }
    }
    __syncthreads();
  }
  if (t < 64) atomicMax(&mEnc[t], f2ord(runmax));
}

// K5: zsum[s] += sum_e exp(score[e][s] - m[s])
__global__ void k_sumexp(const float* __restrict__ x,
                         const float* __restrict__ sec1,
                         const unsigned* __restrict__ mEnc,
                         float* __restrict__ zsum, int E) {
  __shared__ alignas(32) _Float16 sS1[4096];
  __shared__ alignas(32) _Float16 sXT[4096];
  __shared__ float sSc[64 * LDF];
  __shared__ float sm[64];
  int t = threadIdx.x, lane = t & 31, w = t >> 5;
  int mt0 = w >> 2, nt = w & 3;
  int rr = t >> 2, c0 = (t & 3) << 4;
  {
    float v[16];
    load_row16_nc(sec1, rr, c0, v);
    put_a_nat(sS1, rr, c0, v);
  }
  if (t < 64) sm[t] = ord2f(mEnc[t]);
  float runsum = 0.f;
  __syncthreads();
  int nch = (E + 63) >> 6;
  for (int c = blockIdx.x; c < nch; c += gridDim.x) {
    long e0 = (long)c << 6;
    float v[16];
    if (e0 + 64 <= E) load_row16_nc(x, e0 + rr, c0, v);
    else              load_row16(x, e0 + rr, E, c0, v);
    put_b_tr(sXT, rr, c0, v);
    __syncthreads();
#pragma unroll
    for (int tt = 0; tt < 2; ++tt) {
      int mt = mt0 + tt * 2;
      v8f sc = {};
      sc = wmma16(load_frag(sS1, mt * 2 + 0, lane), load_frag(sXT, 0 + nt, lane), sc);
      sc = wmma16(load_frag(sS1, mt * 2 + 1, lane), load_frag(sXT, 4 + nt, lane), sc);
      int nl = lane & 15;
      int mbase = mt * 16 + ((lane & 16) ? 8 : 0);
      bool valid = (e0 + nt * 16 + nl) < E;
#pragma unroll
      for (int i = 0; i < 8; ++i)
        sSc[(mbase + i) * LDF + nt * 16 + nl] = valid ? sc[i] : NEG_BIG;
    }
    __syncthreads();
    if (t < 64) {
      const float4* q = (const float4*)(sSc + t * LDF);
      float m = sm[t];
#pragma unroll
      for (int j = 0; j < 16; ++j) {
        float4 z = q[j];
        runsum += __expf(z.x - m) + __expf(z.y - m) +
                  __expf(z.z - m) + __expf(z.w - m);
      }
    }
    __syncthreads();
  }
  if (t < 64) atomicAdd(&zsum[t], runsum);
}

// K6: p2[s][d] += sum_e exp(score[e][s]-m[s]) * x[e][d]
__global__ void k_sec2_accum(const float* __restrict__ x,
                             const float* __restrict__ sec1,
                             const unsigned* __restrict__ mEnc,
                             float* __restrict__ p2, int E) {
  __shared__ alignas(32) _Float16 sS1[4096];  // A frags: sector1 [s][d]
  __shared__ alignas(32) _Float16 sXT[4096];  // B frags: x^T [d][e]
  __shared__ alignas(32) _Float16 sXN[4096];  // B frags: x [e][d]
  __shared__ alignas(32) _Float16 sW [4096];  // A frags: exp-weights [s][e]
  __shared__ float sm[64];
  int t = threadIdx.x, lane = t & 31, w = t >> 5;
  int mt0 = w >> 2, nt = w & 3;
  int rr = t >> 2, c0 = (t & 3) << 4;
  {
    float v[16];
    load_row16_nc(sec1, rr, c0, v);
    put_a_nat(sS1, rr, c0, v);
  }
  if (t < 64) sm[t] = ord2f(mEnc[t]);
  v8f acc0 = {}, acc1 = {};
  __syncthreads();
  int nch = (E + 63) >> 6;
  for (int c = blockIdx.x; c < nch; c += gridDim.x) {
    long e0 = (long)c << 6;
    float v[16];
    if (e0 + 64 <= E) load_row16_nc(x, e0 + rr, c0, v);
    else              load_row16(x, e0 + rr, E, c0, v);
    put_b_tr(sXT, rr, c0, v);
    put_b_nat(sXN, rr, c0, v);
    __syncthreads();
    // pass 1: adj2^T weight tiles (scores recomputed, exp applied, A frags)
#pragma unroll
    for (int tt = 0; tt < 2; ++tt) {
      int mt = mt0 + tt * 2;
      v8f sc = {};
      sc = wmma16(load_frag(sS1, mt * 2 + 0, lane), load_frag(sXT, 0 + nt, lane), sc);
      sc = wmma16(load_frag(sS1, mt * 2 + 1, lane), load_frag(sXT, 4 + nt, lane), sc);
      int nl = lane & 15;
      int mbase = mt * 16 + ((lane & 16) ? 8 : 0);
      bool valid = (e0 + nt * 16 + nl) < E;
#pragma unroll
      for (int i = 0; i < 8; ++i) {
        int s = mbase + i;
        float wv = valid ? __expf(sc[i] - sm[s]) : 0.f;
        sW[a_addr(s, nt * 16 + nl)] = (_Float16)wv;
      }
    }
    __syncthreads();
    // pass 2: P += W . x_chunk
    acc0 = wmma16(load_frag(sW, mt0 * 2 + 0, lane), load_frag(sXN, 0 + nt, lane), acc0);
    acc0 = wmma16(load_frag(sW, mt0 * 2 + 1, lane), load_frag(sXN, 4 + nt, lane), acc0);
    acc1 = wmma16(load_frag(sW, (mt0 + 2) * 2 + 0, lane), load_frag(sXN, 0 + nt, lane), acc1);
    acc1 = wmma16(load_frag(sW, (mt0 + 2) * 2 + 1, lane), load_frag(sXN, 4 + nt, lane), acc1);
    __syncthreads();
  }
  int n = nt * 16 + (lane & 15);
  int mb = (lane & 16) ? 8 : 0;
#pragma unroll
  for (int i = 0; i < 8; ++i) {
    atomicAdd(&p2[(mt0 * 16 + mb + i) * 64 + n], acc0[i]);
    atomicAdd(&p2[((mt0 + 2) * 16 + mb + i) * 64 + n], acc1[i]);
  }
}

// K7: sector2[s][d] = p2[s][d] / zsum[s]
__global__ void k_sec2_fin(const float* p2, const float* zsum, float* sec2) {
  for (int i = threadIdx.x; i < 4096; i += blockDim.x)
    sec2[i] = p2[i] / zsum[i >> 6];
}

// K8: fused per-entity tail. Each wave owns a 16-entity tile; 48 WMMAs per tile.
__global__ void k_main(const float* __restrict__ x, const float* __restrict__ sec2,
                       const float* __restrict__ Went, const float* __restrict__ bent,
                       const float* __restrict__ Wsi, const float* __restrict__ bsi,
                       const float* __restrict__ Wso, const float* __restrict__ bso,
                       const float* __restrict__ Wou, const float* __restrict__ bou,
                       float* __restrict__ out0, float* __restrict__ out1, int E) {
  __shared__ alignas(32) _Float16 sS2[4096], sS2T[4096];      // B frags
  __shared__ alignas(32) _Float16 sWe[4096], sWi[4096], sWo[4096], sWu[4096];
  __shared__ float sB[4][64];
  __shared__ alignas(16) float sFB[8][16 * LDF];              // per-wave f32 scores
  __shared__ alignas(32) _Float16 sUA[8][1024];               // per-wave f16 A frags
  int t = threadIdx.x, lane = t & 31, w = t >> 5;
  {
    int rr = t >> 2, c0 = (t & 3) << 4;
    float v[16];
    load_row16_nc(sec2, rr, c0, v);
    put_b_nat(sS2, rr, c0, v);                 // B[k=s][n=d] for inv @ sector2
    put_b_tr(sS2T, rr, c0, v);                 // B[k=d][n=s] for x @ sector2^T
    load_row16_nc(Went, rr, c0, v);            // row n=rr, cols k -> B[k][n]=W[n][k]
    put_b_tr(sWe, rr, c0, v);
    load_row16_nc(Wsi, rr, c0, v);
    put_b_tr(sWi, rr, c0, v);
    load_row16_nc(Wso, rr, c0, v);
    put_b_tr(sWo, rr, c0, v);
    load_row16_nc(Wou, rr, c0, v);
    put_b_tr(sWu, rr, c0, v);
  }
  if (t < 64) { sB[0][t] = bent[t]; sB[1][t] = bsi[t]; sB[2][t] = bso[t]; sB[3][t] = bou[t]; }
  __syncthreads();

  float* fb = sFB[w];
  _Float16* ua = sUA[w];
  int r  = lane & 15;
  int kh = (lane & 16) ? 8 : 0;
  int mb = (lane & 16) ? 8 : 0;
  int ntiles = (E + 15) >> 4;                  // E = 500000 -> 31250 exact tiles
  for (int tile = blockIdx.x * 8 + w; tile < ntiles; tile += gridDim.x * 8) {
    long e0 = (long)tile << 4;
    const float* xrow = x + e0 * 64;
    __builtin_prefetch(xrow + (long)gridDim.x * 8 * 16 * 64, 0, 1);

    v16h ax0 = frag_a_gl(xrow, 64, lane, 0);
    v16h ax1 = frag_a_gl(xrow, 64, lane, 32);

    // scores2 = x @ sector2^T   -> fb (f32, real values needed for softmax)
#pragma unroll
    for (int nt = 0; nt < 4; ++nt) {
      v8f c = {};
      c = wmma16(ax0, load_frag(sS2T, 0 + nt, lane), c);
      c = wmma16(ax1, load_frag(sS2T, 4 + nt, lane), c);
      store_d_ldsf(fb, LDF, lane, nt * 16, c);
    }
    // row softmax via float4 LDS accesses (lane pair redundantly owns row r)
    float4* frow = (float4*)(fb + r * LDF);
    float mx = NEG_BIG;
#pragma unroll
    for (int q = 0; q < 16; ++q) {
      float4 z = frow[q];
      mx = fmaxf(mx, fmaxf(fmaxf(z.x, z.y), fmaxf(z.z, z.w)));
    }
    float ssum = 0.f;
#pragma unroll
    for (int q = 0; q < 16; ++q) {
      float4 z = frow[q];
      z.x = __expf(z.x - mx); z.y = __expf(z.y - mx);
      z.z = __expf(z.z - mx); z.w = __expf(z.w - mx);
      ssum += (z.x + z.y) + (z.z + z.w);
      frow[q] = z;
    }
    float isum = 1.f / ssum;
    v16h ia0, ia1;
    {
      int qb = kh >> 2;                        // 0 or 2
#pragma unroll
      for (int blk = 0; blk < 4; ++blk) {
        float4 z0 = frow[blk * 4 + qb];
        float4 z1 = frow[blk * 4 + qb + 1];
        int base = (blk & 1) * 8;
        if (blk < 2) {
          ia0[base + 0] = (_Float16)(z0.x * isum); ia0[base + 1] = (_Float16)(z0.y * isum);
          ia0[base + 2] = (_Float16)(z0.z * isum); ia0[base + 3] = (_Float16)(z0.w * isum);
          ia0[base + 4] = (_Float16)(z1.x * isum); ia0[base + 5] = (_Float16)(z1.y * isum);
          ia0[base + 6] = (_Float16)(z1.z * isum); ia0[base + 7] = (_Float16)(z1.w * isum);
        } else {
          ia1[base + 0] = (_Float16)(z0.x * isum); ia1[base + 1] = (_Float16)(z0.y * isum);
          ia1[base + 2] = (_Float16)(z0.z * isum); ia1[base + 3] = (_Float16)(z0.w * isum);
          ia1[base + 4] = (_Float16)(z1.x * isum); ia1[base + 5] = (_Float16)(z1.y * isum);
          ia1[base + 6] = (_Float16)(z1.z * isum); ia1[base + 7] = (_Float16)(z1.w * isum);
        }
      }
    }
    // ent_update = inv @ sector2 -> ua (f16 A frags)
#pragma unroll
    for (int nt = 0; nt < 4; ++nt) {
      v8f c = {};
      c = wmma16(ia0, load_frag(sS2, 0 + nt, lane), c);
      c = wmma16(ia1, load_frag(sS2, 4 + nt, lane), c);
      store_d_afrag(ua, lane, nt * 16, c);
    }
    v16h ua0 = load_frag(ua, 0, lane);
    v16h ua1 = load_frag(ua, 1, lane);
    // u1 = ent_update @ W_ent^T + b_ent -> ua
#pragma unroll
    for (int nt = 0; nt < 4; ++nt) {
      v8f c = {};
      c = wmma16(ua0, load_frag(sWe, 0 + nt, lane), c);
      c = wmma16(ua1, load_frag(sWe, 4 + nt, lane), c);
      float bv = sB[0][nt * 16 + (lane & 15)];
#pragma unroll
      for (int i = 0; i < 8; ++i) c[i] += bv;
      store_d_afrag(ua, lane, nt * 16, c);
    }
    v16h va0 = load_frag(ua, 0, lane);
    v16h va1 = load_frag(ua, 1, lane);
    // out0 = leaky(u1 @ W_skip_out^T + b_skip_out)
#pragma unroll
    for (int nt = 0; nt < 4; ++nt) {
      v8f c = {};
      c = wmma16(va0, load_frag(sWo, 0 + nt, lane), c);
      c = wmma16(va1, load_frag(sWo, 4 + nt, lane), c);
      int n = nt * 16 + (lane & 15);
      float bv = sB[2][n];
#pragma unroll
      for (int i = 0; i < 8; ++i)
        out0[(e0 + mb + i) * 64 + n] = leaky(c[i] + bv);
    }
    // with_sec = x + u1 @ W_skip_in^T + b_skip_in -> ua
#pragma unroll
    for (int nt = 0; nt < 4; ++nt) {
      v8f c = {};
      c = wmma16(va0, load_frag(sWi, 0 + nt, lane), c);
      c = wmma16(va1, load_frag(sWi, 4 + nt, lane), c);
      int n = nt * 16 + (lane & 15);
      float bv = sB[1][n];
#pragma unroll
      for (int i = 0; i < 8; ++i)
        ua[a16_addr(mb + i, n)] =
            (_Float16)(c[i] + bv + x[(e0 + mb + i) * 64 + n]);
    }
    v16h wa0 = load_frag(ua, 0, lane);
    v16h wa1 = load_frag(ua, 1, lane);
    // out1 = leaky(with_sec @ W_out^T + b_out)
#pragma unroll
    for (int nt = 0; nt < 4; ++nt) {
      v8f c = {};
      c = wmma16(wa0, load_frag(sWu, 0 + nt, lane), c);
      c = wmma16(wa1, load_frag(sWu, 4 + nt, lane), c);
      int n = nt * 16 + (lane & 15);
      float bv = sB[3][n];
#pragma unroll
      for (int i = 0; i < 8; ++i)
        out1[(e0 + mb + i) * 64 + n] = leaky(c[i] + bv);
    }
  }
}

// ---------------------------------------------------------------------------
extern "C" void kernel_launch(void* const* d_in, const int* in_sizes, int n_in,
                              void* d_out, int out_size, void* d_ws, size_t ws_size,
                              hipStream_t stream) {
  (void)n_in; (void)out_size; (void)ws_size;
  const float* x    = (const float*)d_in[0];
  const float* e2s  = (const float*)d_in[1];
  const float* Went = (const float*)d_in[2];
  const float* bent = (const float*)d_in[3];
  const float* Wsi  = (const float*)d_in[4];
  const float* bsi  = (const float*)d_in[5];
  const float* Wso  = (const float*)d_in[6];
  const float* bso  = (const float*)d_in[7];
  const float* Wou  = (const float*)d_in[8];
  const float* bou  = (const float*)d_in[9];
  int E = in_sizes[0] / 64;
  float* out0 = (float*)d_out;                 // ent_skip_out [E,64]
  float* out1 = out0 + (size_t)E * 64;         // ent_to_gnn   [E,64]

  float*    colsum = (float*)d_ws;             // 64
  float*    p1     = colsum + 64;              // 4096
  float*    sec1   = p1 + 4096;                // 4096
  unsigned* mEnc   = (unsigned*)(sec1 + 4096); // 64
  float*    zsum   = (float*)(mEnc + 64);      // 64
  float*    p2     = zsum + 64;                // 4096
  float*    sec2   = p2 + 4096;                // 4096

  k_init<<<1, 256, 0, stream>>>(colsum, p1, mEnc, zsum, p2);
  k_colsum<<<512, 256, 0, stream>>>(e2s, colsum, E);
  k_sec1_accum<<<512, 256, 0, stream>>>(e2s, x, p1, E);
  k_sec1_fin<<<1, 256, 0, stream>>>(p1, colsum, sec1);
  k_colmax<<<512, 256, 0, stream>>>(x, sec1, mEnc, E);
  k_sumexp<<<512, 256, 0, stream>>>(x, sec1, mEnc, zsum, E);
  k_sec2_accum<<<512, 256, 0, stream>>>(x, sec1, mEnc, p2, E);
  k_sec2_fin<<<1, 256, 0, stream>>>(p2, zsum, sec2);
  int tiles = (E + 15) >> 4;
  int blocks = (tiles + 7) / 8;
  k_main<<<blocks, 256, 0, stream>>>(x, sec2, Went, bent, Wsi, bsi, Wso, bso,
                                     Wou, bou, out0, out1, E);
}
